// CausalSelfAttention_75342316306516
// MI455X (gfx1250) — matrix operand order repaired and checked by reading, not compiled, not run on hardware
//
#include <hip/hip_runtime.h>
#include <hip/hip_bf16.h>
#include <math.h>

// Problem constants (fixed by the reference)
#define BB     4
#define T_SEQ  2048
#define CC     1024
#define NHEAD  16
#define HD     64     // CC / NHEAD

typedef __attribute__((ext_vector_type(16))) __bf16          v16bf;
typedef __attribute__((ext_vector_type(8)))  float           v8f;
typedef __attribute__((ext_vector_type(8)))  unsigned short  ush8;
typedef __attribute__((ext_vector_type(4)))  unsigned short  ush4;

union FragAB {
    v16bf v;
    ush8  h[2];
};

__device__ __forceinline__ unsigned short f2bf(float f) {
    union { float f; unsigned int u; } c; c.f = f;
    unsigned int u = c.u;
    return (unsigned short)((u + 0x7FFFu + ((u >> 16) & 1u)) >> 16);
}

// ---------------------------------------------------------------------------
// f32 -> bf16 cast, 4 elements / thread
// ---------------------------------------------------------------------------
__global__ __launch_bounds__(256) void cast_f32_bf16(
    const float* __restrict__ src, unsigned short* __restrict__ dst, int n4)
{
    int i = blockIdx.x * blockDim.x + threadIdx.x;
    if (i < n4) {
        float4 f = ((const float4*)src)[i];
        ush4 o;
        o.x = f2bf(f.x); o.y = f2bf(f.y); o.z = f2bf(f.z); o.w = f2bf(f.w);
        ((ush4*)dst)[i] = o;
    }
}

// ---------------------------------------------------------------------------
// WMMA GEMM:  out[m,n] = sum_k A[m,k] * W[n,k] + bias[n]
//   A: bf16 [M,K] row-major,  W: bf16 [N,K] row-major
//   MODE 0: store bf16 to [B,H,T,hd]   (m=b*T+t, n=h*hd+d)  -> Q,K
//   MODE 2: store bf16 to [B,H,hd,T]   (V transposed)
//   MODE 3: store f32  to [M,N] row-major (final projection -> d_out)
// 128x128 block tile, 256 threads (8 waves), 32x64 per wave, BK=32.
// Staging: GLOBAL_LOAD_ASYNC_TO_LDS_B128 (ASYNCcnt), double-buffered LDS so
// tile k+1's DMA overlaps tile k's 8 WMMAs; s_wait_asynccnt 4 pipelines.
// ---------------------------------------------------------------------------
template <int MODE>
__global__ __launch_bounds__(256) void gemm_nt(
    const unsigned short* __restrict__ A,
    const unsigned short* __restrict__ W,
    const float* __restrict__ bias,
    unsigned short* __restrict__ outb,
    float* __restrict__ outf,
    int M, int N, int K)
{
    constexpr int BM = 128, BN = 128, BK = 32;
    __shared__ __align__(16) unsigned short As[2][BM][40];  // pad: conflict-free b128
    __shared__ __align__(16) unsigned short Bs[2][BN][40];

    const int tid   = threadIdx.x;
    const int lane  = tid & 31;
    const int wave  = tid >> 5;
    const int wm0   = (wave & 3) * 32;   // 4 waves along M
    const int wn0   = (wave >> 2) * 64;  // 2 waves along N
    const int m0    = blockIdx.y * BM;
    const int n0    = blockIdx.x * BN;
    const int laneM = lane & 15;
    const int laneH = lane >> 4;
    const int klo   = laneH * 8;         // A-operand chunk base
    const int khalf = laneH * 16;        // unused; B reads also use klo

    // async-copy one 128x32 A tile + 128x32 W tile into LDS buffer `buf`
    // (4 x b128 per thread, memory -> LDS, no VGPR data, tracked by ASYNCcnt)
    auto issue_tile = [&](int buf, int k0) {
#pragma unroll
        for (int i = 0; i < 2; i++) {
            int c   = tid + i * 256;
            int row = c >> 2;
            int seg = (c & 3) * 8;
            unsigned lA = (unsigned)(uintptr_t)&As[buf][row][seg];
            unsigned lB = (unsigned)(uintptr_t)&Bs[buf][row][seg];
            unsigned long long gA =
                (unsigned long long)(uintptr_t)(A + (size_t)(m0 + row) * K + k0 + seg);
            unsigned long long gB =
                (unsigned long long)(uintptr_t)(W + (size_t)(n0 + row) * K + k0 + seg);
            asm volatile("global_load_async_to_lds_b128 %0, %1, off"
                         :: "v"(lA), "v"(gA) : "memory");
            asm volatile("global_load_async_to_lds_b128 %0, %1, off"
                         :: "v"(lB), "v"(gB) : "memory");
        }
    };

    v8f acc[2][4];
#pragma unroll
    for (int i = 0; i < 2; i++)
#pragma unroll
        for (int j = 0; j < 4; j++) acc[i][j] = {};

    const int nK = K / BK;
    issue_tile(0, 0);

    for (int kt = 0; kt < nK; kt++) {
        const int buf = kt & 1;
        if (kt + 1 < nK) {
            issue_tile(buf ^ 1, (kt + 1) * BK);          // overlap next tile's DMA
            asm volatile("s_wait_asynccnt 0x4" ::: "memory");  // current tile done
        } else {
            asm volatile("s_wait_asynccnt 0x0" ::: "memory");  // drain
        }
        __syncthreads();   // all waves' async tiles visible in LDS

        // --- per-wave fragments from LDS ---
        FragAB af[2], bfr[4];
#pragma unroll
        for (int mi = 0; mi < 2; mi++) {
            int r = wm0 + mi * 16 + laneM;
            af[mi].h[0] = *(const ush8*)(&As[buf][r][klo]);
            af[mi].h[1] = *(const ush8*)(&As[buf][r][16 + klo]);
        }
#pragma unroll
        for (int ni = 0; ni < 4; ni++) {
            int r = wn0 + ni * 16 + laneM;
            bfr[ni].h[0] = *(const ush8*)(&Bs[buf][r][klo]);
            bfr[ni].h[1] = *(const ush8*)(&Bs[buf][r][16 + klo]);
        }
#pragma unroll
        for (int mi = 0; mi < 2; mi++)
#pragma unroll
            for (int ni = 0; ni < 4; ni++)
                acc[mi][ni] = __builtin_amdgcn_wmma_f32_16x16x32_bf16(
                    false, af[mi].v, false, bfr[ni].v,
                    (short)0, acc[mi][ni], false, false);
        __syncthreads();   // everyone done reading `buf` before it is re-filled
    }

    // --- epilogue: bias add + layout-specific store ---
#pragma unroll
    for (int mi = 0; mi < 2; mi++) {
#pragma unroll
        for (int ni = 0; ni < 4; ni++) {
            int nIdx = n0 + wn0 + ni * 16 + laneM;
            float bv = bias[nIdx];
#pragma unroll
            for (int v = 0; v < 8; v++) {
                int mIdx  = m0 + wm0 + mi * 16 + v + laneH * 8;
                float val = acc[mi][ni][v] + bv;
                if (MODE == 3) {
                    outf[(size_t)mIdx * N + nIdx] = val;
                } else {
                    int b = mIdx >> 11, t = mIdx & (T_SEQ - 1);
                    int h = nIdx >> 6,  d = nIdx & (HD - 1);
                    size_t idx;
                    if (MODE == 0)
                        idx = ((((size_t)b * NHEAD + h) * T_SEQ) + t) * HD + d;
                    else
                        idx = ((((size_t)b * NHEAD + h) * HD) + d) * T_SEQ + t;
                    outb[idx] = f2bf(val);
                }
            }
        }
    }
}

// ---------------------------------------------------------------------------
// Flash attention (causal + key/query padding mask), one wave per 16 q-rows.
//   Q,K: bf16 [B,H,T,hd];  Vt: bf16 [B,H,hd,T];  Y: bf16 [B,T,H,hd]
// 32 keys/iter: 4 WMMA for S (=Q·Kt), online softmax, 4 WMMA for P·V.
// ---------------------------------------------------------------------------
__global__ __launch_bounds__(128) void attn_kernel(
    const unsigned short* __restrict__ Q,
    const unsigned short* __restrict__ Km,
    const unsigned short* __restrict__ Vt,
    const unsigned char*  __restrict__ amask,
    unsigned short* __restrict__ Y)
{
    __shared__ __align__(16) unsigned short pT[4][16 * 32];  // per-wave P tile

    const int lane  = threadIdx.x & 31;
    const int wave  = threadIdx.x >> 5;
    const int h     = blockIdx.y;
    const int b     = blockIdx.z;
    const int q0    = (blockIdx.x * 4 + wave) * 16;
    const int laneM = lane & 15;
    const int laneH = lane >> 4;

    const size_t headOff = ((size_t)b * NHEAD + h) * T_SEQ * HD;
    const unsigned short* Qh = Q  + headOff;
    const unsigned short* Kh = Km + headOff;
    const unsigned short* Vh = Vt + headOff;   // [hd][T]

    // Q fragments (kept in registers for the whole key sweep)
    FragAB qf[2];
    {
        const unsigned short* qr = Qh + (size_t)(q0 + laneM) * HD;
#pragma unroll
        for (int di = 0; di < 2; di++) {
            qf[di].h[0] = *(const ush8*)(qr + di * 32 + laneH * 8);
            qf[di].h[1] = *(const ush8*)(qr + di * 32 + 16 + laneH * 8);
        }
    }

    // per-lane row state; slot v corresponds to row  q0 + v + laneH*8
    bool  qmask[8];
    float rmax[8], rsum[8];
    v8f   o[4];
#pragma unroll
    for (int ni = 0; ni < 4; ni++) o[ni] = {};
#pragma unroll
    for (int v = 0; v < 8; v++) {
        int qi   = q0 + v + laneH * 8;
        qmask[v] = amask[(size_t)b * T_SEQ + qi] != 0;
        rmax[v]  = -1e30f;
        rsum[v]  = 0.f;
    }

    const float SCALE = 0.125f;  // 1/sqrt(64)
    const int kTiles  = (q0 + 16 + 31) >> 5;   // causal limit

    for (int kt = 0; kt < kTiles; kt++) {
        const int k0 = kt * 32;

        // ---- S = Q · K^T for 2 key sub-tiles of 16 ----
        v8f s[2];
        bool kmask[2];
#pragma unroll
        for (int j = 0; j < 2; j++) {
            const int key = k0 + j * 16 + laneM;
            kmask[j] = amask[(size_t)b * T_SEQ + key] != 0;
            const unsigned short* kr = Kh + (size_t)key * HD;
            v8f sj = {};
#pragma unroll
            for (int di = 0; di < 2; di++) {
                FragAB kf;
                kf.h[0] = *(const ush8*)(kr + di * 32 + laneH * 8);
                kf.h[1] = *(const ush8*)(kr + di * 32 + 16 + laneH * 8);
                sj = __builtin_amdgcn_wmma_f32_16x16x32_bf16(
                    false, qf[di].v, false, kf.v, (short)0, sj, false, false);
            }
            s[j] = sj;
        }

        // ---- mask + scale, local row-max ----
        float mloc[8];
#pragma unroll
        for (int v = 0; v < 8; v++) {
            int qi = q0 + v + laneH * 8;
            int kA = k0 + laneM, kBk = k0 + 16 + laneM;
            float s0 = s[0][v] * SCALE, s1 = s[1][v] * SCALE;
            bool ok0 = qmask[v] && kmask[0] && (kA  <= qi);
            bool ok1 = qmask[v] && kmask[1] && (kBk <= qi);
            s0 = ok0 ? s0 : -1e9f;
            s1 = ok1 ? s1 : -1e9f;
            s[0][v] = s0; s[1][v] = s1;
            mloc[v] = fmaxf(s0, s1);
        }
#pragma unroll
        for (int off = 1; off < 16; off <<= 1)
#pragma unroll
            for (int v = 0; v < 8; v++)
                mloc[v] = fmaxf(mloc[v], __shfl_xor(mloc[v], off, 32));

        // ---- online softmax update ----
        float corr[8], pl[8];
#pragma unroll
        for (int v = 0; v < 8; v++) {
            float nm = fmaxf(rmax[v], mloc[v]);
            corr[v]  = __expf(rmax[v] - nm);
            rmax[v]  = nm;
            float p0 = __expf(s[0][v] - nm);
            float p1 = __expf(s[1][v] - nm);
            s[0][v] = p0; s[1][v] = p1;
            pl[v] = p0 + p1;
        }
#pragma unroll
        for (int ni = 0; ni < 4; ni++)
#pragma unroll
            for (int v = 0; v < 8; v++) o[ni][v] *= corr[v];
#pragma unroll
        for (int off = 1; off < 16; off <<= 1)
#pragma unroll
            for (int v = 0; v < 8; v++)
                pl[v] += __shfl_xor(pl[v], off, 32);
#pragma unroll
        for (int v = 0; v < 8; v++) rsum[v] = rsum[v] * corr[v] + pl[v];

        // ---- P (C-layout f32) -> A-operand bf16 via per-wave LDS transpose ----
        unsigned short* pw = pT[wave];
#pragma unroll
        for (int j = 0; j < 2; j++)
#pragma unroll
            for (int v = 0; v < 8; v++) {
                int m  = v + laneH * 8;
                int kk = j * 16 + laneM;
                pw[m * 32 + kk] = f2bf(s[j][v]);
            }
        asm volatile("s_wait_dscnt 0" ::: "memory");  // DS stores complete (CDNA5 split cnt)
        FragAB pf;
        {
            const unsigned short* pr = pw + laneM * 32;
            pf.h[0] = *(const ush8*)(pr + laneH * 8);
            pf.h[1] = *(const ush8*)(pr + 16 + laneH * 8);
        }

        // ---- O += P · V  (V^T rows are contiguous in t) ----
#pragma unroll
        for (int ni = 0; ni < 4; ni++) {
            const unsigned short* vr =
                Vh + (size_t)(ni * 16 + laneM) * T_SEQ + k0 + laneH * 8;
            FragAB vf;
            vf.h[0] = *(const ush8*)(vr);
            vf.h[1] = *(const ush8*)(vr + 16);
            o[ni] = __builtin_amdgcn_wmma_f32_16x16x32_bf16(
                false, pf.v, false, vf.v, (short)0, o[ni], false, false);
        }
    }

    // ---- normalize and store Y[b,t,h,d] ----
#pragma unroll
    for (int v = 0; v < 8; v++) rsum[v] = 1.f / rsum[v];
#pragma unroll
    for (int ni = 0; ni < 4; ni++)
#pragma unroll
        for (int v = 0; v < 8; v++) {
            int t = q0 + v + laneH * 8;
            int d = ni * 16 + laneM;
            size_t idx = (((size_t)b * T_SEQ + t) * NHEAD + h) * HD + d;
            Y[idx] = f2bf(o[ni][v] * rsum[v]);
        }
}

// ---------------------------------------------------------------------------
// launch
// ---------------------------------------------------------------------------
extern "C" void kernel_launch(void* const* d_in, const int* in_sizes, int n_in,
                              void* d_out, int out_size, void* d_ws, size_t ws_size,
                              hipStream_t stream) {
    (void)in_sizes; (void)n_in; (void)out_size; (void)ws_size;
    const float*         x  = (const float*)d_in[0];
    const unsigned char* am = (const unsigned char*)d_in[1];  // jnp bool = 1 byte
    const float* Wq = (const float*)d_in[2];
    const float* bq = (const float*)d_in[3];
    const float* Wk = (const float*)d_in[4];
    const float* bk = (const float*)d_in[5];
    const float* Wv = (const float*)d_in[6];
    const float* bv = (const float*)d_in[7];
    const float* Wp = (const float*)d_in[8];
    const float* bp = (const float*)d_in[9];

    const int M  = BB * T_SEQ;            // 8192
    const int nX = M * CC;                // 8,388,608
    const int nW = CC * CC;               // 1,048,576

    unsigned short* ws = (unsigned short*)d_ws;
    size_t off = 0;
    auto take = [&](size_t n) { unsigned short* p = ws + off; off += n; return p; };
    unsigned short* xb  = take(nX);
    unsigned short* wqb = take(nW);
    unsigned short* wkb = take(nW);
    unsigned short* wvb = take(nW);
    unsigned short* wpb = take(nW);
    unsigned short* qb  = take(nX);   // [B,H,T,hd]
    unsigned short* kb  = take(nX);
    unsigned short* vtb = take(nX);   // [B,H,hd,T]
    unsigned short* yb  = take(nX);   // [B,T,H,hd]

    // casts
    cast_f32_bf16<<<nX / 4 / 256, 256, 0, stream>>>(x,  xb,  nX / 4);
    cast_f32_bf16<<<nW / 4 / 256, 256, 0, stream>>>(Wq, wqb, nW / 4);
    cast_f32_bf16<<<nW / 4 / 256, 256, 0, stream>>>(Wk, wkb, nW / 4);
    cast_f32_bf16<<<nW / 4 / 256, 256, 0, stream>>>(Wv, wvb, nW / 4);
    cast_f32_bf16<<<nW / 4 / 256, 256, 0, stream>>>(Wp, wpb, nW / 4);

    dim3 ggrid(CC / 128, M / 128);  // (8, 64)
    gemm_nt<0><<<ggrid, 256, 0, stream>>>(xb, wqb, bq, qb,  nullptr, M, CC, CC);
    gemm_nt<0><<<ggrid, 256, 0, stream>>>(xb, wkb, bk, kb,  nullptr, M, CC, CC);
    gemm_nt<2><<<ggrid, 256, 0, stream>>>(xb, wvb, bv, vtb, nullptr, M, CC, CC);

    attn_kernel<<<dim3(T_SEQ / 64, NHEAD, BB), 128, 0, stream>>>(qb, kb, vtb, am, yb);

    gemm_nt<3><<<ggrid, 256, 0, stream>>>(yb, wpb, bp, nullptr, (float*)d_out, M, CC, CC);
}
